// EDRNCell_51659866637126
// MI455X (gfx1250) — matrix operand order repaired and checked by reading, compiled, not run on hardware
//
#include <hip/hip_runtime.h>
#include <math.h>

typedef __attribute__((ext_vector_type(16))) _Float16 v16h;
typedef __attribute__((ext_vector_type(8)))  float    v8f;
typedef unsigned int u32x4 __attribute__((ext_vector_type(4)));
typedef int          i32x8 __attribute__((ext_vector_type(8)));
typedef int          i32x4 __attribute__((ext_vector_type(4)));

#define B_ROWS 1024
#define N_IN   256
#define M_DIM  512
#define D_DIM  4
#define MD     2048

__device__ __forceinline__ float sigf(float x) { return 1.0f / (1.0f + __expf(-x)); }

// ---------------------------------------------------------------------------
// TDM: DMA a 2-D fp32 tile [n_rows x n_cols] (row pitch = pitch elements) from
// global memory into LDS (row-major, contiguous). D# layout per CDNA5 ISA §8.
//   group0: count=1 | lds_addr | global_addr[56:0] | type=2
//   group1: wg_mask=0, data_size=2 (4B), tensor dims/strides, tile dims
//   group2/3/extra: zero (tile_dim2=0 -> unused; 2-D tensor)
// This toolchain exposes the 6-arg builtin form.
// ---------------------------------------------------------------------------
__device__ __forceinline__ void tdm_load_2d(const float* gsrc, unsigned lds_byte_off,
                                            int n_cols, int n_rows, int pitch) {
  unsigned long long ga = (unsigned long long)(uintptr_t)gsrc;
  u32x4 g0;
  g0[0] = 1u;                                              // count=1, user mode
  g0[1] = lds_byte_off;                                    // bits 63:32  lds_addr
  g0[2] = (unsigned)(ga & 0xffffffffu);                    // bits 95:64  global_addr lo
  g0[3] = (unsigned)((ga >> 32) & 0x1ffffffu) | (2u << 30);// addr hi + type=2
  i32x8 g1;
  g1[0] = (int)(2u << 16);                                 // data_size=2 (4 bytes)
  g1[1] = (int)(((unsigned)pitch & 0xffffu) << 16);        // tensor_dim0[15:0]
  g1[2] = (int)((((unsigned)pitch >> 16) & 0xffffu)        // tensor_dim0[31:16]
              | (((unsigned)n_rows & 0xffffu) << 16));     // tensor_dim1[15:0]
  g1[3] = (int)(((unsigned)n_cols & 0xffffu) << 16);       // tile_dim0 (bits 127:112)
  g1[4] = (int)((unsigned)n_rows & 0xffffu);               // tile_dim1 ; tile_dim2=0
  g1[5] = pitch;                                           // tensor_dim0_stride[31:0]
  g1[6] = 0;                                               // stride hi / dim1_stride
  g1[7] = 0;
  i32x4 gz  = {0, 0, 0, 0};
  i32x8 gz8 = {0, 0, 0, 0, 0, 0, 0, 0};
  __builtin_amdgcn_tensor_load_to_lds(g0, g1, gz, gz, gz8, 0);
}

__device__ __forceinline__ unsigned lds_off(const void* p) {
  return (unsigned)(uintptr_t)p;   // low 32 bits of a shared-aperture address = LDS byte offset
}

// A-fragment: 16x32 f16, rows rb..rb+15, K = kb..kb+31, converted from fp32.
// (estride/eoff read the strided "last slice" view a[:, :, D-1] in place.)
__device__ __forceinline__ v16h load_a_frag(const float* __restrict__ src, int ld,
                                            int rb, int kb, int estride, int eoff) {
  const int lane = threadIdx.x & 31;
  const int row  = rb + (lane & 15);
  const int kk   = kb + ((lane >> 4) << 3);
  const float* p = src + (size_t)row * ld;
  v16h a;
#pragma unroll
  for (int j = 0; j < 8; ++j) a[j]     = (_Float16)p[(kk + j) * estride + eoff];
#pragma unroll
  for (int j = 0; j < 8; ++j) a[8 + j] = (_Float16)p[(kk + 16 + j) * estride + eoff];
  return a;
}

// B-fragment from an LDS-resident 32x16 fp32 tile (row-major, TDM-filled).
// lane<16 -> col=lane, K 0..7,16..23 ; lane>=16 -> col=lane-16, K 8..15,24..31.
// For fixed j, the 16 lanes of a half read 16 consecutive dwords -> no bank conflict.
__device__ __forceinline__ v16h b_frag_lds(const float* sW) {
  const int lane = threadIdx.x & 31;
  const int c  = lane & 15;
  const int k0 = (lane >> 4) << 3;
  v16h b;
#pragma unroll
  for (int j = 0; j < 8; ++j) b[j] = (_Float16)sW[(k0 + j) * 16 + c];
#pragma unroll
  for (int j = 0; j < 8; ++j) b[8 + j] = (_Float16)sW[(k0 + 16 + j) * 16 + c];
  return b;
}

#define WMMA_F16(acc, a, b) \
  acc = __builtin_amdgcn_wmma_f32_16x16x32_f16(false, (a), false, (b), (short)0, (acc), false, false)

// ---------------- Kernel 1: G_fg, G_in, G_th (+ sparse A_pt) -> a_new ----------------
// Workgroup = 8 waves stacked in M: 128 rows x 16 cols of output; one TDM weight
// tile per gate per K-step feeds all 8 waves.
__global__ void __launch_bounds__(256) k_gates1(
    const float* __restrict__ x_t, const float* __restrict__ m_t,
    const float* __restrict__ a_t,
    const float* __restrict__ A_fg, const float* __restrict__ A_in,
    const float* __restrict__ A_th, const float* __restrict__ A_pt,
    const float* __restrict__ B_fg, const float* __restrict__ B_in,
    const float* __restrict__ B_th,
    const float* __restrict__ b_fg, const float* __restrict__ b_in,
    const float* __restrict__ b_th,
    float* __restrict__ a_new) {
  __shared__ float sW[3][32 * 16];
  const int wave = threadIdx.x >> 5;
  const int lane = threadIdx.x & 31;
  const int rb = ((blockIdx.x >> 7) << 7) + (wave << 4);   // 8 row-groups of 128
  const int cb = (blockIdx.x & 127) << 4;                  // 128 col tiles

  v8f acc_fg = {}; v8f acc_in = {}; v8f acc_th = {};

  // K over M: m_t@A_fg, m_t@A_in, a_last@A_th
  for (int kb = 0; kb < M_DIM; kb += 32) {
    if (wave == 0) {
      tdm_load_2d(A_fg + (size_t)kb * MD + cb, lds_off(&sW[0][0]), 16, 32, MD);
      tdm_load_2d(A_in + (size_t)kb * MD + cb, lds_off(&sW[1][0]), 16, 32, MD);
      tdm_load_2d(A_th + (size_t)kb * MD + cb, lds_off(&sW[2][0]), 16, 32, MD);
      __builtin_amdgcn_s_wait_tensorcnt(0);
    }
    __syncthreads();
    v16h am = load_a_frag(m_t, M_DIM, rb, kb, 1, 0);
    v16h al = load_a_frag(a_t, MD,    rb, kb, D_DIM, D_DIM - 1);
    v16h wfg = b_frag_lds(sW[0]);
    v16h win = b_frag_lds(sW[1]);
    v16h wth = b_frag_lds(sW[2]);
    WMMA_F16(acc_fg, am, wfg);
    WMMA_F16(acc_in, am, win);
    WMMA_F16(acc_th, al, wth);
    __syncthreads();
  }
  // K over N: x_t @ B_{fg,in,th}
  for (int kb = 0; kb < N_IN; kb += 32) {
    if (wave == 0) {
      tdm_load_2d(B_fg + (size_t)kb * MD + cb, lds_off(&sW[0][0]), 16, 32, MD);
      tdm_load_2d(B_in + (size_t)kb * MD + cb, lds_off(&sW[1][0]), 16, 32, MD);
      tdm_load_2d(B_th + (size_t)kb * MD + cb, lds_off(&sW[2][0]), 16, 32, MD);
      __builtin_amdgcn_s_wait_tensorcnt(0);
    }
    __syncthreads();
    v16h ax = load_a_frag(x_t, N_IN, rb, kb, 1, 0);
    v16h wfg = b_frag_lds(sW[0]);
    v16h win = b_frag_lds(sW[1]);
    v16h wth = b_frag_lds(sW[2]);
    WMMA_F16(acc_fg, ax, wfg);
    WMMA_F16(acc_in, ax, win);
    WMMA_F16(acc_th, ax, wth);
    __syncthreads();
  }

  const int col = cb + (lane & 15);
  const int hi  = lane >> 4;
  const float bias_fg = b_fg[col], bias_in = b_in[col], bias_th = b_th[col];

  // Sparse a_t @ (A_pt*pt_mask): only (j*D+s, j*D+s2), s<s2, survive the mask.
  const int s2 = col & (D_DIM - 1);
  const int jb = col & ~(D_DIM - 1);
  float w0 = 0.f, w1 = 0.f, w2 = 0.f;
  if (s2 > 0) w0 = A_pt[(size_t)(jb + 0) * MD + col];
  if (s2 > 1) w1 = A_pt[(size_t)(jb + 1) * MD + col];
  if (s2 > 2) w2 = A_pt[(size_t)(jb + 2) * MD + col];

#pragma unroll
  for (int v = 0; v < 8; ++v) {
    const int row = rb + v + (hi << 3);
    const float* arow = a_t + (size_t)row * MD;
    float pt = 0.f;
    if (s2 > 0) pt += arow[jb + 0] * w0;
    if (s2 > 1) pt += arow[jb + 1] * w1;
    if (s2 > 2) pt += arow[jb + 2] * w2;
    const float gfg = sigf(acc_fg[v] + bias_fg);
    const float gin = sigf(acc_in[v] + bias_in);
    const float gth = tanhf(acc_th[v] + bias_th + pt);
    a_new[(size_t)row * MD + col] = arow[col] * gfg + gth * gin;
  }
}

// ---------------- Kernel 2: G_ot, aa, m_partial, aa_last ----------------
__global__ void __launch_bounds__(256) k_gates2(
    const float* __restrict__ x_t, const float* __restrict__ a_new,
    const float* __restrict__ A_ot, const float* __restrict__ B_ot,
    const float* __restrict__ b_ot,
    float* __restrict__ aa_last, float* __restrict__ m_partial) {
  __shared__ float sW[32 * 16];
  const int wave = threadIdx.x >> 5;
  const int lane = threadIdx.x & 31;
  const int rb = ((blockIdx.x >> 7) << 7) + (wave << 4);
  const int cb = (blockIdx.x & 127) << 4;

  v8f acc = {};
  for (int kb = 0; kb < M_DIM; kb += 32) {       // a_new_last @ A_ot
    if (wave == 0) {
      tdm_load_2d(A_ot + (size_t)kb * MD + cb, lds_off(&sW[0]), 16, 32, MD);
      __builtin_amdgcn_s_wait_tensorcnt(0);
    }
    __syncthreads();
    v16h al = load_a_frag(a_new, MD, rb, kb, D_DIM, D_DIM - 1);
    v16h w  = b_frag_lds(sW);
    WMMA_F16(acc, al, w);
    __syncthreads();
  }
  for (int kb = 0; kb < N_IN; kb += 32) {        // x_t @ B_ot
    if (wave == 0) {
      tdm_load_2d(B_ot + (size_t)kb * MD + cb, lds_off(&sW[0]), 16, 32, MD);
      __builtin_amdgcn_s_wait_tensorcnt(0);
    }
    __syncthreads();
    v16h ax = load_a_frag(x_t, N_IN, rb, kb, 1, 0);
    v16h w  = b_frag_lds(sW);
    WMMA_F16(acc, ax, w);
    __syncthreads();
  }

  const int col = cb + (lane & 15);
  const int hi  = lane >> 4;
  const float bias = b_ot[col];

#pragma unroll
  for (int v = 0; v < 8; ++v) {
    const int row = rb + v + (hi << 3);
    const float an = a_new[(size_t)row * MD + col];
    const float aa = tanhf(an) * sigf(acc[v] + bias);
    // reduce across the D=4 columns of one j-block (lane bits 0..1)
    float t  = aa + __shfl_xor(aa, 1, 32);
    float s4 = t + __shfl_xor(t, 2, 32);
    if ((col & 3) == 3) {
      const int j = col >> 2;
      aa_last[(size_t)row * M_DIM + j]   = aa;        // s = D-1 slice
      m_partial[(size_t)row * M_DIM + j] = s4 - aa;   // sum over s < D-1
    }
  }
}

// ---------------- Kernel 3: m_new = m_partial + aa_last @ A_st ----------------
__global__ void __launch_bounds__(256) k_mnew(
    const float* __restrict__ aa_last, const float* __restrict__ A_st,
    const float* __restrict__ m_partial, float* __restrict__ m_new) {
  __shared__ float sW[32 * 16];
  const int wave = threadIdx.x >> 5;
  const int lane = threadIdx.x & 31;
  const int rb = ((blockIdx.x >> 5) << 7) + (wave << 4);   // 8 row-groups of 128
  const int cb = (blockIdx.x & 31) << 4;                   // 32 col tiles

  v8f acc = {};
  for (int kb = 0; kb < M_DIM; kb += 32) {
    if (wave == 0) {
      tdm_load_2d(A_st + (size_t)kb * M_DIM + cb, lds_off(&sW[0]), 16, 32, M_DIM);
      __builtin_amdgcn_s_wait_tensorcnt(0);
    }
    __syncthreads();
    v16h a = load_a_frag(aa_last, M_DIM, rb, kb, 1, 0);
    v16h w = b_frag_lds(sW);
    WMMA_F16(acc, a, w);
    __syncthreads();
  }

  const int col = cb + (lane & 15);
  const int hi  = lane >> 4;
#pragma unroll
  for (int v = 0; v < 8; ++v) {
    const int row = rb + v + (hi << 3);
    m_new[(size_t)row * M_DIM + col] = acc[v] + m_partial[(size_t)row * M_DIM + col];
  }
}

extern "C" void kernel_launch(void* const* d_in, const int* in_sizes, int n_in,
                              void* d_out, int out_size, void* d_ws, size_t ws_size,
                              hipStream_t stream) {
  const float* x_t  = (const float*)d_in[0];
  const float* m_t  = (const float*)d_in[1];
  const float* a_t  = (const float*)d_in[2];
  const float* A_fg = (const float*)d_in[3];
  const float* A_in = (const float*)d_in[4];
  const float* A_th = (const float*)d_in[5];
  const float* A_ot = (const float*)d_in[6];
  const float* A_st = (const float*)d_in[7];
  const float* A_pt = (const float*)d_in[8];
  const float* B_fg = (const float*)d_in[9];
  const float* B_in = (const float*)d_in[10];
  const float* B_th = (const float*)d_in[11];
  const float* B_ot = (const float*)d_in[12];
  const float* b_fg = (const float*)d_in[13];
  const float* b_in = (const float*)d_in[14];
  const float* b_th = (const float*)d_in[15];
  const float* b_ot = (const float*)d_in[16];
  (void)in_sizes; (void)n_in; (void)out_size; (void)ws_size;

  float* out_m = (float*)d_out;                      // [1024, 512]
  float* out_a = (float*)d_out + B_ROWS * M_DIM;     // [1024, 2048]
  float* ws_aa_last   = (float*)d_ws;                                // [1024, 512]
  float* ws_m_partial = (float*)d_ws + (size_t)B_ROWS * M_DIM;       // [1024, 512]

  // [B, MD] kernels: 8 row-groups x 128 col-tiles = 1024 blocks of 8 waves.
  k_gates1<<<1024, 256, 0, stream>>>(x_t, m_t, a_t, A_fg, A_in, A_th, A_pt,
                                     B_fg, B_in, B_th, b_fg, b_in, b_th, out_a);
  k_gates2<<<1024, 256, 0, stream>>>(x_t, out_a, A_ot, B_ot, b_ot,
                                     ws_aa_last, ws_m_partial);
  // [B, M]: 8 row-groups x 32 col-tiles = 256 blocks.
  k_mnew<<<256, 256, 0, stream>>>(ws_aa_last, A_st, ws_m_partial, out_m);
}
